// MeshAttention_35983236006201
// MI455X (gfx1250) — matrix operand order repaired and checked by reading, not compiled, not run on hardware
//
#include <hip/hip_runtime.h>
#include <cstdint>
#include <cstddef>

// ---------------- problem constants ----------------
#define B_SZ   4
#define E_SZ   1500
#define EP_SZ  1504          // E padded to 16
#define NQT    94            // ceil(E/16)
#define D_MODEL 256
#define N_HEADS 8
#define DK_SZ  32
#define DV_SZ  32
#define PMAXV  5
#define MAXDIST 3
#define ROWS   (B_SZ * E_SZ) // 6000
#define MASKBIT 0x40

typedef __attribute__((ext_vector_type(16))) __bf16 v16bf;
typedef __attribute__((ext_vector_type(8)))  float  v8f;

static __device__ __forceinline__ v8f v8f_zero() {
  v8f z;
#pragma unroll
  for (int i = 0; i < 8; ++i) z[i] = 0.0f;
  return z;
}

// A fragment, 16-bit 16x32 (MxK). Lane m (and m+16) hold row m.
// lanes 0-15: K = 0..15; lanes 16-31: K = 16..31.
static __device__ __forceinline__ v16bf load_a16(const __bf16* __restrict__ rowp, int laneh) {
  v16bf a;
  const int base = laneh * 16;
#pragma unroll
  for (int i = 0; i < 16; ++i) a[i] = rowp[base + i];
  return a;
}

// B fragment, 16-bit 32x16 (KxN). Lane n holds column n:
// lanes 0-15: K=0..15, lanes 16-31: K=16..31 (caller adds laneh*16 to rowp).
static __device__ __forceinline__ v16bf load_b16(const __bf16* __restrict__ rowp) {
  v16bf b;
#pragma unroll
  for (int i = 0; i < 16; ++i) b[i] = rowp[i];
  return b;
}

static __device__ __forceinline__ v8f wmma_bf16(v16bf a, v16bf b, v8f c) {
  return __builtin_amdgcn_wmma_f32_16x16x32_bf16(false, a, false, b, (short)0, c, false, false);
}

// ---------------- weight transpose + bf16 convert: WT[n][k] = W[k][n] ----------------
__global__ void wt_kernel(const float* __restrict__ W, __bf16* __restrict__ WT) {
  int n = blockIdx.x * 16 + threadIdx.x;
  int k = blockIdx.y * 16 + threadIdx.y;
  WT[(size_t)n * 256 + k] = (__bf16)W[(size_t)k * 256 + n];
}

// ---------------- dist -> packed u8 code: MASKBIT if d>3, low bits = min(d,5) ----------------
__global__ __launch_bounds__(256) void didx_kernel(const int* __restrict__ dist,
                                                   unsigned char* __restrict__ bidx) {
  const int i = blockIdx.x * 256 + threadIdx.x;       // index over groups of 4
  if (i >= (B_SZ * E_SZ * E_SZ) / 4) return;
  const int4 d = ((const int4*)dist)[i];
  uchar4 o;
  o.x = (unsigned char)(((d.x > MAXDIST) ? MASKBIT : 0) | (d.x > PMAXV ? PMAXV : d.x));
  o.y = (unsigned char)(((d.y > MAXDIST) ? MASKBIT : 0) | (d.y > PMAXV ? PMAXV : d.y));
  o.z = (unsigned char)(((d.z > MAXDIST) ? MASKBIT : 0) | (d.z > PMAXV ? PMAXV : d.z));
  o.w = (unsigned char)(((d.w > MAXDIST) ? MASKBIT : 0) | (d.w > PMAXV ? PMAXV : d.w));
  ((uchar4*)bidx)[i] = o;
}

// ---------------- layernorm (+ raw copy) to bf16, [B,E,D] row-major ----------------
__global__ __launch_bounds__(256) void ln_kernel(const float* __restrict__ x,
                                                 const float* __restrict__ w,
                                                 const float* __restrict__ bvec,
                                                 __bf16* __restrict__ s_bf,
                                                 __bf16* __restrict__ qn_bf) {
  __shared__ float red[256];
  const int be = blockIdx.x;          // b*E + e
  const int b  = be / E_SZ, e = be % E_SZ;
  const int d  = threadIdx.x;
  const float v = x[((size_t)b * D_MODEL + d) * E_SZ + e];
  red[d] = v; __syncthreads();
  for (int s = 128; s > 0; s >>= 1) { if (d < s) red[d] += red[d + s]; __syncthreads(); }
  const float mu = red[0] * (1.0f / 256.0f);
  __syncthreads();
  const float c = v - mu;
  red[d] = c * c; __syncthreads();
  for (int s = 128; s > 0; s >>= 1) { if (d < s) red[d] += red[d + s]; __syncthreads(); }
  const float var = red[0] * (1.0f / 256.0f);
  const float y = c * rsqrtf(var + 1e-6f) * w[d] + bvec[d];
  s_bf [(size_t)be * 256 + d] = (__bf16)v;
  qn_bf[(size_t)be * 256 + d] = (__bf16)y;
}

// ---------------- projection GEMM: [6000,256](bf16) x WT[256,256] -> bf16 ----------------
// mode 0: out row-major [6000,256]; mode 1: out = vT layout [B,H,DV,E]
__global__ __launch_bounds__(128) void proj_gemm(const __bf16* __restrict__ A,
                                                 const __bf16* __restrict__ WT,
                                                 __bf16* __restrict__ out,
                                                 float scale, int mode) {
  const int rt = blockIdx.x;                 // 0..374
  const int wave = threadIdx.x >> 5, lane = threadIdx.x & 31;
  const int lanelow = lane & 15, laneh = lane >> 4;
  const __bf16* Abase = A + ((size_t)rt * 16 + lanelow) * 256;

  v8f acc[4];
#pragma unroll
  for (int j = 0; j < 4; ++j) acc[j] = v8f_zero();

#pragma unroll
  for (int kk = 0; kk < 8; ++kk) {
    __builtin_prefetch(Abase + (kk + 1) * 32, 0, 3);
    v16bf a = load_a16(Abase + kk * 32, laneh);
#pragma unroll
    for (int j = 0; j < 4; ++j) {
      const int n = (wave * 4 + j) * 16 + lanelow;
      v16bf bf = load_b16(WT + (size_t)n * 256 + kk * 32 + laneh * 16);
      acc[j] = wmma_bf16(a, bf, acc[j]);
    }
  }

#pragma unroll
  for (int j = 0; j < 4; ++j) {
    const int col = (wave * 4 + j) * 16 + lanelow;     // C: N = lane%16
#pragma unroll
    for (int r = 0; r < 8; ++r) {
      const int row_m = rt * 16 + r + (laneh << 3);    // C: M = r + 8*(lane>=16)
      const float vv = acc[j][r] * scale;
      if (mode == 0) {
        out[(size_t)row_m * 256 + col] = (__bf16)vv;
      } else {
        const int b = row_m / E_SZ, e = row_m % E_SZ;
        const int h = col >> 5, dch = col & 31;
        out[(((size_t)b * N_HEADS + h) * DV_SZ + dch) * E_SZ + e] = (__bf16)vv;
      }
    }
  }
}

// ---------------- fc GEMM with residual, writes x_out [B,D,E] fp32 ----------------
__global__ __launch_bounds__(128) void fc_gemm(const __bf16* __restrict__ A,
                                               const __bf16* __restrict__ WT,
                                               const float* __restrict__ xin,
                                               float* __restrict__ xout) {
  const int rt = blockIdx.x;
  const int wave = threadIdx.x >> 5, lane = threadIdx.x & 31;
  const int lanelow = lane & 15, laneh = lane >> 4;
  const __bf16* Abase = A + ((size_t)rt * 16 + lanelow) * 256;

  v8f acc[4];
#pragma unroll
  for (int j = 0; j < 4; ++j) acc[j] = v8f_zero();

#pragma unroll
  for (int kk = 0; kk < 8; ++kk) {
    v16bf a = load_a16(Abase + kk * 32, laneh);
#pragma unroll
    for (int j = 0; j < 4; ++j) {
      const int n = (wave * 4 + j) * 16 + lanelow;
      v16bf bf = load_b16(WT + (size_t)n * 256 + kk * 32 + laneh * 16);
      acc[j] = wmma_bf16(a, bf, acc[j]);
    }
  }

#pragma unroll
  for (int j = 0; j < 4; ++j) {
    const int col = (wave * 4 + j) * 16 + lanelow;
#pragma unroll
    for (int r = 0; r < 8; ++r) {
      const int row_m = rt * 16 + r + (laneh << 3);
      const int b = row_m / E_SZ, e = row_m % E_SZ;
      const size_t o = ((size_t)b * D_MODEL + col) * E_SZ + e;
      xout[o] = acc[j][r] + xin[o];
    }
  }
}

// ---------------- fused attention: QK^T -> bias/mask -> softmax -> attn out + A*V ----------------
__global__ __launch_bounds__(128) void attn_kernel(const __bf16* __restrict__ qbf,  // [B*E,256] (scaled)
                                                   const __bf16* __restrict__ kbf,  // [B*E,256]
                                                   const __bf16* __restrict__ vT,   // [B,H,DV,E]
                                                   const unsigned char* __restrict__ bidx, // [B,E,E]
                                                   const float* __restrict__ rpr,   // [6,32]
                                                   float* __restrict__ attn_out,    // [B,H,E,E]
                                                   float* __restrict__ attn_sum,    // [B,E]
                                                   __bf16* __restrict__ out_heads)  // [B*E,256]
{
  extern __shared__ float smem[];
  float* logits = smem;                     // 16 * EP_SZ
  float* bias   = logits + 16 * EP_SZ;      // 16 * 8
  float* red    = bias + 16 * 8;            // 4 * 16 * 32
  float* rowinv = red + 4 * 16 * 32;        // 16

  const int blk = blockIdx.x;
  const int qt = blk % NQT;
  const int bh = blk / NQT;
  const int h = bh % N_HEADS, b = bh / N_HEADS;
  const int q0 = qt * 16;
  const int tid = threadIdx.x;
  const int wave = tid >> 5, lane = tid & 31;
  const int lanelow = lane & 15, laneh = lane >> 4;

  // relative-position bias p[r][j] = qs[q] . rpr[j]
  if (tid < 96) {
    const int r = tid / 6, j = tid % 6;
    int q = q0 + r; if (q >= E_SZ) q = E_SZ - 1;
    const __bf16* qp = qbf + ((size_t)b * E_SZ + q) * 256 + h * 32;
    float s = 0.0f;
#pragma unroll
    for (int d = 0; d < 32; ++d) s += (float)qp[d] * rpr[j * 32 + d];
    bias[r * 8 + j] = s;
  }

  // Q fragment (K = DK = 32 -> single WMMA depth)
  int qrow = q0 + lanelow; if (qrow >= E_SZ) qrow = E_SZ - 1;
  const v16bf aq = load_a16(qbf + ((size_t)b * E_SZ + qrow) * 256 + h * 32, laneh);

  // logits = qs @ k^T, striped over waves
  for (int kt = wave; kt < NQT; kt += 4) {
    int kcol = kt * 16 + lanelow; if (kcol >= E_SZ) kcol = E_SZ - 1;
    v16bf bk = load_b16(kbf + ((size_t)b * E_SZ + kcol) * 256 + h * 32 + laneh * 16);
    v8f c = v8f_zero();
    c = wmma_bf16(aq, bk, c);
#pragma unroll
    for (int r = 0; r < 8; ++r)
      logits[(r + (laneh << 3)) * EP_SZ + kt * 16 + lanelow] = c[r];
  }
  __syncthreads();

  // + gathered RPR bias, distance mask (packed u8 codes)
  for (int r = 0; r < 16; ++r) {
    const int q = q0 + r;
    if (q < E_SZ) {
      const unsigned char* dp = bidx + ((size_t)b * E_SZ + q) * E_SZ;
      for (int c = tid; c < E_SZ; c += 128) {
        const unsigned int v = dp[c];
        float l = logits[r * EP_SZ + c] + bias[r * 8 + (v & 7)];
        if (v & MASKBIT) l = -1e9f;
        logits[r * EP_SZ + c] = l;
      }
    } else {
      for (int c = tid; c < E_SZ; c += 128) logits[r * EP_SZ + c] = -1e9f;
    }
  }
  __syncthreads();

  // softmax stats: 8 lanes per row
  const int r = tid >> 3, sub = tid & 7;
  float mx = -3.0e38f;
  for (int c = sub; c < E_SZ; c += 8) mx = fmaxf(mx, logits[r * EP_SZ + c]);
#pragma unroll
  for (int o = 4; o > 0; o >>= 1) mx = fmaxf(mx, __shfl_xor(mx, o, 8));
  float sum = 0.0f;
  for (int c = sub; c < E_SZ; c += 8) {
    const float pe = __expf(logits[r * EP_SZ + c] - mx);
    logits[r * EP_SZ + c] = pe;
    sum += pe;
  }
#pragma unroll
  for (int o = 4; o > 0; o >>= 1) sum += __shfl_xor(sum, o, 8);
  if (sub == 0) {
    rowinv[r] = 1.0f / sum;
    for (int c = E_SZ; c < EP_SZ; ++c) logits[r * EP_SZ + c] = 0.0f;  // pad for A*V
  }
  __syncthreads();

  // normalize + stream attn out, full block per row (coalesced, non-temporal)
  for (int rr = 0; rr < 16; ++rr) {
    const float invv = rowinv[rr];
    float* lrow = logits + rr * EP_SZ;
    const int q = q0 + rr;
    if (q < E_SZ) {
      float* arow = attn_out + (((size_t)b * N_HEADS + h) * E_SZ + q) * E_SZ;
      for (int c = tid; c < E_SZ; c += 128) {
        const float p = lrow[c] * invv;
        lrow[c] = p;
        __builtin_nontemporal_store(p, arow + c);
      }
    } else {
      for (int c = tid; c < E_SZ; c += 128) lrow[c] *= invv;
    }
  }
  __syncthreads();

  // per-edge masked sums (over this block's h and 16 q rows)
  for (int c = tid; c < E_SZ; c += 128) {
    float sacc = 0.0f;
    for (int rr = 0; rr < 16; ++rr) {
      const int q = q0 + rr;
      if (q >= E_SZ) break;
      if (!(bidx[((size_t)b * E_SZ + q) * E_SZ + c] & MASKBIT)) sacc += logits[rr * EP_SZ + c];
    }
    atomicAdd(attn_sum + b * E_SZ + c, sacc);
  }

  // out = attn @ v  (A from LDS probabilities, B from transposed V)
  v8f c0 = v8f_zero(), c1 = v8f_zero();
  const __bf16* vbase = vT + ((size_t)b * N_HEADS + h) * DV_SZ * E_SZ;
  for (int kc = wave; kc < EP_SZ / 32; kc += 4) {
    v16bf a;
    const int kb = kc * 32 + laneh * 16;
#pragma unroll
    for (int i = 0; i < 16; ++i) a[i] = (__bf16)logits[lanelow * EP_SZ + kb + i];
    v16bf b0, b1;
    const int kkb = kc * 32 + laneh * 16;
#pragma unroll
    for (int i = 0; i < 16; ++i) {
      int kk = kkb + i; if (kk >= E_SZ) kk = E_SZ - 1;   // attn pad is 0, value harmless
      b0[i] = vbase[(size_t)(lanelow)      * E_SZ + kk];
      b1[i] = vbase[(size_t)(16 + lanelow) * E_SZ + kk];
    }
    c0 = wmma_bf16(a, b0, c0);
    c1 = wmma_bf16(a, b1, c1);
  }
  float* myred = red + wave * 512;
#pragma unroll
  for (int rr = 0; rr < 8; ++rr) {
    const int m = rr + (laneh << 3);
    myred[m * 32 + lanelow]      = c0[rr];
    myred[m * 32 + 16 + lanelow] = c1[rr];
  }
  __syncthreads();
  for (int idx = tid; idx < 512; idx += 128) {
    const float s = red[idx] + red[512 + idx] + red[1024 + idx] + red[1536 + idx];
    const int m = idx >> 5, n = idx & 31;
    const int q = q0 + m;
    if (q < E_SZ) out_heads[((size_t)b * E_SZ + q) * 256 + h * 32 + n] = (__bf16)s;
  }
}

// ---------------- attn_per_edge = attn_sum / count(mask over q) ----------------
__global__ __launch_bounds__(128) void edge_kernel(const unsigned char* __restrict__ bidx,
                                                   const float* __restrict__ attn_sum,
                                                   float* __restrict__ ape) {
  __shared__ int red[128];
  const int bk = blockIdx.x;
  const int b = bk / E_SZ, k = bk % E_SZ;
  int cnt = 0;
  for (int q = threadIdx.x; q < E_SZ; q += 128)
    cnt += (bidx[((size_t)b * E_SZ + q) * E_SZ + k] & MASKBIT) ? 0 : 1;
  red[threadIdx.x] = cnt; __syncthreads();
  for (int s = 64; s > 0; s >>= 1) { if (threadIdx.x < s) red[threadIdx.x] += red[threadIdx.x + s]; __syncthreads(); }
  if (threadIdx.x == 0) ape[bk] = attn_sum[bk] / (float)red[0];
}

// ---------------- launcher ----------------
extern "C" void kernel_launch(void* const* d_in, const int* in_sizes, int n_in,
                              void* d_out, int out_size, void* d_ws, size_t ws_size,
                              hipStream_t stream) {
  const float* x    = (const float*)d_in[0];
  const int*   dist = (const int*)  d_in[1];
  const float* Wq   = (const float*)d_in[2];
  const float* Wk   = (const float*)d_in[3];
  const float* Wv   = (const float*)d_in[4];
  const float* Wfc  = (const float*)d_in[5];
  const float* lnw  = (const float*)d_in[6];
  const float* lnb  = (const float*)d_in[7];
  const float* rpr  = (const float*)d_in[8];

  float* xout = (float*)d_out;                                  // [B,D,E]
  float* attn = xout + (size_t)B_SZ * D_MODEL * E_SZ;           // [B,H,E,E]
  float* ape  = attn + (size_t)B_SZ * N_HEADS * E_SZ * E_SZ;    // [B,E]

  char* ws = (char*)d_ws;
  const size_t act = (size_t)ROWS * 256 * sizeof(__bf16);       // 3,072,000 B each
  __bf16* qn  = (__bf16*)ws; ws += act;
  __bf16* sb  = (__bf16*)ws; ws += act;
  __bf16* qb  = (__bf16*)ws; ws += act;
  __bf16* kb  = (__bf16*)ws; ws += act;
  __bf16* vt  = (__bf16*)ws; ws += act;
  __bf16* oh  = (__bf16*)ws; ws += act;
  __bf16* WqT = (__bf16*)ws; ws += 256 * 256 * sizeof(__bf16);
  __bf16* WkT = (__bf16*)ws; ws += 256 * 256 * sizeof(__bf16);
  __bf16* WvT = (__bf16*)ws; ws += 256 * 256 * sizeof(__bf16);
  __bf16* WfT = (__bf16*)ws; ws += 256 * 256 * sizeof(__bf16);
  float* asum = (float*)ws;  ws += (size_t)ROWS * sizeof(float);
  unsigned char* bidx = (unsigned char*)ws; ws += (size_t)B_SZ * E_SZ * E_SZ;

  const dim3 tb16(16, 16), gb16(16, 16);
  wt_kernel<<<gb16, tb16, 0, stream>>>(Wq,  WqT);
  wt_kernel<<<gb16, tb16, 0, stream>>>(Wk,  WkT);
  wt_kernel<<<gb16, tb16, 0, stream>>>(Wv,  WvT);
  wt_kernel<<<gb16, tb16, 0, stream>>>(Wfc, WfT);

  const int ngrp4 = (B_SZ * E_SZ * E_SZ) / 4;
  didx_kernel<<<(ngrp4 + 255) / 256, 256, 0, stream>>>(dist, bidx);

  ln_kernel<<<ROWS, 256, 0, stream>>>(x, lnw, lnb, sb, qn);

  proj_gemm<<<ROWS / 16, 128, 0, stream>>>(qn, WqT, qb, 0.17677669529663687f, 0); // 1/sqrt(32)
  proj_gemm<<<ROWS / 16, 128, 0, stream>>>(sb, WkT, kb, 1.0f, 0);
  proj_gemm<<<ROWS / 16, 128, 0, stream>>>(sb, WvT, vt, 1.0f, 1);

  hipMemsetAsync(asum, 0, (size_t)ROWS * sizeof(float), stream);

  const size_t smem_bytes = (size_t)(16 * EP_SZ + 16 * 8 + 4 * 16 * 32 + 16) * sizeof(float); // ~105 KB
  attn_kernel<<<B_SZ * N_HEADS * NQT, 128, smem_bytes, stream>>>(qb, kb, vt, bidx, rpr,
                                                                 attn, asum, oh);

  fc_gemm<<<ROWS / 16, 128, 0, stream>>>(oh, WfT, x, xout);

  edge_kernel<<<ROWS, 128, 0, stream>>>(bidx, asum, ape);
}
